// DeltaNet_22488448762145
// MI455X (gfx1250) — compile-verified, matrix operand order checked by reading, not run on hardware
//
#include <hip/hip_runtime.h>

#define Bq   4
#define Lq   4096
#define HID  1024
#define NH   8
#define DH   128
#define NTOK (Bq * Lq)          // 16384
#define CHUNK 32                // scan LDS chunk
#define TILE  128               // GEMM block tile (M and N)
#define KS    32                // GEMM K-slab per WMMA
#define LDSTR 40                // padded LDS row stride (elements) = 80B

typedef __attribute__((ext_vector_type(16))) __bf16 v16bf;
typedef __attribute__((ext_vector_type(8)))  __bf16 v8bf;
typedef __attribute__((ext_vector_type(8)))  float  v8f;

// ---------- bf16 helpers on raw ushort storage ----------
__device__ __forceinline__ unsigned short f2bf(float f) {
    unsigned u = __float_as_uint(f);
    unsigned r = (u + 0x7FFFu + ((u >> 16) & 1u)) >> 16;   // round-to-nearest-even
    return (unsigned short)r;
}
__device__ __forceinline__ float bf2f(unsigned short u) {
    return __uint_as_float(((unsigned)u) << 16);
}

// Build a 16-lane-pattern WMMA fragment from a padded LDS tile.
// lane&15 = row within 16-row tile; lanes>=16 use K offset +8; halves at +0,+16.
__device__ __forceinline__ v16bf lds_frag(const unsigned short* __restrict__ rowbase,
                                          int lane) {
    const unsigned short* p = rowbase + (size_t)(lane & 15) * LDSTR + ((lane >> 4) ? 8 : 0);
    v8bf lo = *(const v8bf*)(p);
    v8bf hi = *(const v8bf*)(p + 16);
    return __builtin_shufflevector(lo, hi, 0,1,2,3,4,5,6,7,8,9,10,11,12,13,14,15);
}

// ---------- 1) RMSNorm: f32 x -> bf16 h ----------
__global__ __launch_bounds__(256)
void rmsnorm_kernel(const float* __restrict__ x, const float* __restrict__ w,
                    unsigned short* __restrict__ h) {
    const int token = blockIdx.x;
    const float* xp = x + (size_t)token * HID;
    float ss = 0.f;
    for (int i = threadIdx.x; i < HID; i += 256) { float v = xp[i]; ss += v * v; }
    #pragma unroll
    for (int o = 16; o; o >>= 1) ss += __shfl_xor(ss, o, 32);
    __shared__ float red[8];
    __shared__ float rinv;
    if ((threadIdx.x & 31) == 0) red[threadIdx.x >> 5] = ss;
    __syncthreads();
    if (threadIdx.x == 0) {
        float t = 0.f;
        #pragma unroll
        for (int i = 0; i < 8; ++i) t += red[i];
        rinv = rsqrtf(t * (1.0f / HID) + 1e-6f);
    }
    __syncthreads();
    const float rv = rinv;
    for (int i = threadIdx.x; i < HID; i += 256)
        h[(size_t)token * HID + i] = f2bf(xp[i] * rv * w[i]);
}

// ---------- 2) f32 -> bf16 weight convert ----------
__global__ __launch_bounds__(256)
void cvt_bf16_kernel(const float* __restrict__ a, unsigned short* __restrict__ o, int n) {
    int i = blockIdx.x * 256 + threadIdx.x;
    if (i < n) o[i] = f2bf(a[i]);
}

// ---------- 3) gate kernel: sigmoid(h . W[g] + b) per head ----------
__global__ __launch_bounds__(256)
void gates_kernel(const unsigned short* __restrict__ h,
                  const float* __restrict__ Wb, const float* __restrict__ bb,
                  const float* __restrict__ Wi, const float* __restrict__ bi,
                  const float* __restrict__ Wo, const float* __restrict__ bo,
                  float* __restrict__ gb, float* __restrict__ gi, float* __restrict__ go) {
    const int token = blockIdx.x;
    const int lane = threadIdx.x & 31, head = threadIdx.x >> 5;  // 8 waves = 8 heads
    const unsigned short* hp = h + (size_t)token * HID;
    const float* Ws[3]  = {Wb, Wi, Wo};
    const float* bs[3]  = {bb, bi, bo};
    float*       out[3] = {gb, gi, go};
    #pragma unroll
    for (int g = 0; g < 3; ++g) {
        const float* wrow = Ws[g] + (size_t)head * HID;
        float s = 0.f;
        for (int d = lane; d < HID; d += 32) s += bf2f(hp[d]) * wrow[d];
        #pragma unroll
        for (int o = 16; o; o >>= 1) s += __shfl_xor(s, o, 32);
        if (lane == 0) {
            float z = s + bs[g][head];
            out[g][(size_t)token * NH + head] = 1.f / (1.f + __expf(-z));
        }
    }
}

// ---------- 4) WMMA GEMM: C[M,N] = A[M,K](bf16) * W[N,K]^T(bf16) (+bias) ----------
// 128x128 block tile, 8 waves (4M x 2N), each wave 32x64 = 8 wmma tiles.
// LDS double-buffered; 64 wmma per 16KB staged slab.
__global__ __launch_bounds__(256)
void gemm_bf16_wmma(const unsigned short* __restrict__ A,
                    const unsigned short* __restrict__ W,
                    const float* __restrict__ bias,
                    float* __restrict__ C, int M, int N, int K) {
    __shared__ unsigned short As[2][TILE * LDSTR];
    __shared__ unsigned short Bs[2][TILE * LDSTR];

    const int tid  = threadIdx.x;
    const int lane = tid & 31;
    const int wave = tid >> 5;            // 0..7
    const int wm = wave >> 1;             // 0..3 : 32-row group
    const int wn = wave & 1;              // 0..1 : 64-col group

    const int bm = blockIdx.x * TILE;
    const int bn = blockIdx.y * TILE;

    // staging: each thread owns 16 contiguous bf16 (two b128) per matrix
    const int srow = tid >> 1;            // 0..127
    const int scol = (tid & 1) * 16;      // 0 or 16
    const unsigned short* gA = A + (size_t)(bm + srow) * K + scol;
    const unsigned short* gB = W + (size_t)(bn + srow) * K + scol;
    const int sidx = srow * LDSTR + scol;

    v8f acc[2][4] = {};

    // preload slab 0 and stage it
    uint4 ra0 = *(const uint4*)(gA);
    uint4 ra1 = *(const uint4*)(gA + 8);
    uint4 rb0 = *(const uint4*)(gB);
    uint4 rb1 = *(const uint4*)(gB + 8);
    *(uint4*)(&As[0][sidx])     = ra0;
    *(uint4*)(&As[0][sidx + 8]) = ra1;
    *(uint4*)(&Bs[0][sidx])     = rb0;
    *(uint4*)(&Bs[0][sidx + 8]) = rb1;
    __syncthreads();

    const int nslab = K / KS;
    for (int s = 0; s < nslab; ++s) {
        const int cur = s & 1;
        // issue global loads for slab s+1 (consumed after compute)
        if (s + 1 < nslab) {
            const unsigned short* pA = gA + (s + 1) * KS;
            const unsigned short* pB = gB + (s + 1) * KS;
            ra0 = *(const uint4*)(pA);
            ra1 = *(const uint4*)(pA + 8);
            rb0 = *(const uint4*)(pB);
            rb1 = *(const uint4*)(pB + 8);
            if (s + 2 < nslab) {               // hint slab s+2 -> global_prefetch_b8
                __builtin_prefetch(pA + KS, 0, 1);
                __builtin_prefetch(pB + KS, 0, 1);
            }
        }

        // fragments from LDS
        v16bf afr[2], bfr[4];
        #pragma unroll
        for (int i = 0; i < 2; ++i)
            afr[i] = lds_frag(&As[cur][(wm * 32 + i * 16) * LDSTR], lane);
        #pragma unroll
        for (int j = 0; j < 4; ++j)
            bfr[j] = lds_frag(&Bs[cur][(wn * 64 + j * 16) * LDSTR], lane);

        #pragma unroll
        for (int i = 0; i < 2; ++i)
            #pragma unroll
            for (int j = 0; j < 4; ++j)
                acc[i][j] = __builtin_amdgcn_wmma_f32_16x16x32_bf16(
                    false, afr[i], false, bfr[j], (short)0, acc[i][j], false, false);

        if (s + 1 < nslab) {
            const int nxt = cur ^ 1;   // other buffer: safe to fill while peers compute
            *(uint4*)(&As[nxt][sidx])     = ra0;
            *(uint4*)(&As[nxt][sidx + 8]) = ra1;
            *(uint4*)(&Bs[nxt][sidx])     = rb0;
            *(uint4*)(&Bs[nxt][sidx + 8]) = rb1;
            __syncthreads();
        }
    }

    // writeback: VGPR r -> M = r (+8 for lanes 16-31), N = lane&15
    const int colb = bn + wn * 64 + (lane & 15);
    const int rowb = bm + wm * 32 + ((lane >> 4) ? 8 : 0);
    #pragma unroll
    for (int j = 0; j < 4; ++j) {
        const int col = colb + j * 16;
        const float bv = bias ? bias[col] : 0.f;
        #pragma unroll
        for (int i = 0; i < 2; ++i) {
            #pragma unroll
            for (int r = 0; r < 8; ++r)
                C[(size_t)(rowb + i * 16 + r) * N + col] = acc[i][j][r] + bv;
        }
    }
}

// ---------- 5) RoPE in-place on q and k (f32) ----------
__global__ __launch_bounds__(256)
void rope_kernel(float* __restrict__ q, float* __restrict__ k) {
    size_t idx = (size_t)blockIdx.x * 256 + threadIdx.x;     // over B*L*NH*64
    if (idx >= (size_t)Bq * Lq * NH * (DH / 2)) return;
    int i = idx & 63; size_t r = idx >> 6;
    int hd = r & 7; r >>= 3;
    int l = (int)(r & (Lq - 1)); int b = (int)(r >> 12);
    float inv = __powf(10000.f, -(float)i * (1.0f / (DH / 2)));
    float ang = (float)l * inv;
    float s, c; __sincosf(ang, &s, &c);
    size_t base = (((size_t)b * Lq + l) * HID) + (size_t)hd * DH + i;
    float q1 = q[base], q2 = q[base + 64];
    q[base] = q1 * c - q2 * s;  q[base + 64] = q1 * s + q2 * c;
    float k1 = k[base], k2 = k[base + 64];
    k[base] = k1 * c - k2 * s;  k[base + 64] = k1 * s + k2 * c;
}

// ---------- 6) gated delta scan ----------
// Columns of M are independent: one wave32 block owns 32 columns of one (b,h).
// grid = B*NH*4 blocks; lane holds its column (128 f32 in VGPRs).
__global__ __launch_bounds__(32)
void scan_kernel(const float* __restrict__ q, const float* __restrict__ k,
                 const float* __restrict__ v,
                 const float* __restrict__ gb, const float* __restrict__ gi,
                 const float* __restrict__ go,
                 unsigned short* __restrict__ ybf) {
    const int blk = blockIdx.x;
    const int cg = blk & 3;                    // column group (32 cols)
    const int bh = blk >> 2;
    const int h = bh & (NH - 1), b = bh >> 3;
    const int e = cg * 32 + threadIdx.x;       // owned column of M

    __shared__ float lq[CHUNK * DH], lk[CHUNK * DH], lv[CHUNK * 32];
    __shared__ float lf[CHUNK], lg[CHUNK], lo[CHUNK];

    float m[DH];
    #pragma unroll
    for (int d = 0; d < DH; ++d) m[d] = 0.f;
    const float scale = 0.08838834764831845f;  // 1/sqrt(128)

    for (int t0 = 0; t0 < Lq; t0 += CHUNK) {
        for (int idx = threadIdx.x; idx < CHUNK * DH; idx += 32) {
            int t = idx >> 7, d = idx & (DH - 1);
            size_t g = (((size_t)b * Lq + t0 + t) * HID) + (size_t)h * DH + d;
            lq[idx] = q[g]; lk[idx] = k[g];
        }
        for (int idx = threadIdx.x; idx < CHUNK * 32; idx += 32) {
            int t = idx >> 5, d = idx & 31;
            lv[idx] = v[(((size_t)b * Lq + t0 + t) * HID) + (size_t)h * DH + cg * 32 + d];
        }
        {   // CHUNK == 32 == blockDim
            size_t gg = ((size_t)b * Lq + t0 + threadIdx.x) * NH + h;
            lf[threadIdx.x] = gb[gg]; lg[threadIdx.x] = gi[gg]; lo[threadIdx.x] = go[gg];
        }
        __syncthreads();

        for (int t = 0; t < CHUNK; ++t) {
            const float f = lf[t], g = lg[t] * scale, o = lo[t];
            const float gv = g * lv[t * 32 + threadIdx.x];
            float acc = 0.f;
            #pragma unroll
            for (int d = 0; d < DH; ++d) {
                float md = fmaf(m[d], f, gv * lk[t * DH + d]);   // lk/lq: LDS broadcast
                m[d] = md;
                acc = fmaf(lq[t * DH + d], md, acc);
            }
            size_t yi = (((size_t)b * Lq + t0 + t) * HID) + (size_t)h * DH + e;
            ybf[yi] = f2bf(acc * o);
        }
        __syncthreads();
    }
}

// ---------------- host launcher ----------------
extern "C" void kernel_launch(void* const* d_in, const int* in_sizes, int n_in,
                              void* d_out, int out_size, void* d_ws, size_t ws_size,
                              hipStream_t stream) {
    const float* x     = (const float*)d_in[0];
    const float* normw = (const float*)d_in[1];
    const float* Wq    = (const float*)d_in[2];
    const float* Wk    = (const float*)d_in[3];
    const float* Wv    = (const float*)d_in[4];
    const float* Wbeta = (const float*)d_in[5];
    const float* bbeta = (const float*)d_in[6];
    const float* Wig   = (const float*)d_in[7];
    const float* big   = (const float*)d_in[8];
    const float* Wog   = (const float*)d_in[9];
    const float* bog   = (const float*)d_in[10];
    const float* Wout  = (const float*)d_in[11];
    const float* bout  = (const float*)d_in[12];
    float* out = (float*)d_out;

    char* ws = (char*)d_ws;
    auto alloc = [&](size_t bytes) {
        void* p = (void*)ws;
        ws += (bytes + 255) & ~(size_t)255;
        return p;
    };
    unsigned short* hbf    = (unsigned short*)alloc((size_t)NTOK * HID * 2);
    unsigned short* Wq_bf  = (unsigned short*)alloc((size_t)HID * HID * 2);
    unsigned short* Wk_bf  = (unsigned short*)alloc((size_t)HID * HID * 2);
    unsigned short* Wv_bf  = (unsigned short*)alloc((size_t)HID * HID * 2);
    unsigned short* Wo_bf  = (unsigned short*)alloc((size_t)HID * HID * 2);
    float*          qb     = (float*)alloc((size_t)NTOK * HID * 4);
    float*          kb     = (float*)alloc((size_t)NTOK * HID * 4);
    float*          vb     = (float*)alloc((size_t)NTOK * HID * 4);
    unsigned short* ybf    = (unsigned short*)alloc((size_t)NTOK * HID * 2);
    float*          g_beta = (float*)alloc((size_t)NTOK * NH * 4);
    float*          g_ig   = (float*)alloc((size_t)NTOK * NH * 4);
    float*          g_og   = (float*)alloc((size_t)NTOK * NH * 4);
    (void)ws_size; (void)in_sizes; (void)n_in; (void)out_size;

    // 1) RMSNorm
    rmsnorm_kernel<<<NTOK, 256, 0, stream>>>(x, normw, hbf);

    // 2) weight conversion to bf16
    const int nW = HID * HID, gW = (nW + 255) / 256;
    cvt_bf16_kernel<<<gW, 256, 0, stream>>>(Wq,   Wq_bf, nW);
    cvt_bf16_kernel<<<gW, 256, 0, stream>>>(Wk,   Wk_bf, nW);
    cvt_bf16_kernel<<<gW, 256, 0, stream>>>(Wv,   Wv_bf, nW);
    cvt_bf16_kernel<<<gW, 256, 0, stream>>>(Wout, Wo_bf, nW);

    // 3) gates
    gates_kernel<<<NTOK, 256, 0, stream>>>(hbf, Wbeta, bbeta, Wig, big, Wog, bog,
                                           g_beta, g_ig, g_og);

    // 4) q,k,v projections via WMMA (128x128 tiles)
    dim3 gg(NTOK / TILE, HID / TILE);
    gemm_bf16_wmma<<<gg, 256, 0, stream>>>(hbf, Wq_bf, nullptr, qb, NTOK, HID, HID);
    gemm_bf16_wmma<<<gg, 256, 0, stream>>>(hbf, Wk_bf, nullptr, kb, NTOK, HID, HID);
    gemm_bf16_wmma<<<gg, 256, 0, stream>>>(hbf, Wv_bf, nullptr, vb, NTOK, HID, HID);

    // 5) RoPE on q,k
    const size_t nRope = (size_t)Bq * Lq * NH * (DH / 2);
    rope_kernel<<<(unsigned)((nRope + 255) / 256), 256, 0, stream>>>(qb, kb);

    // 6) recurrent scan -> y (bf16); 4 column-group blocks per (b,h)
    scan_kernel<<<Bq * NH * 4, 32, 0, stream>>>(qb, kb, vb, g_beta, g_ig, g_og, ybf);

    // 7) output projection + bias via WMMA
    gemm_bf16_wmma<<<gg, 256, 0, stream>>>(ybf, Wo_bf, bout, out, NTOK, HID, HID);
}
